// TransformerEncoderBlock_2482491097269
// MI455X (gfx1250) — matrix operand order repaired and checked by reading, not compiled, not run on hardware
//
#include <hip/hip_runtime.h>
#include <cstdint>

// ---------------------------------------------------------------------------
// Transformer encoder block for MI455X (gfx1250), wave32 + WMMA bf16.
// - All matmuls on v_wmma_f32_16x16x32_bf16 (f32 accumulate).
// - Weights converted+transposed to bf16 once per launch.
// - LDS staging uses gfx1250 GLOBAL_LOAD_ASYNC_TO_LDS_B128 (ASYNCcnt) with
//   double-buffered tiles: one barrier per K-step, copies overlap WMMA.
// ---------------------------------------------------------------------------

#define S_LEN 2048
#define E_DIM 1024
#define H_NUM 16
#define HD_DIM 64
#define MLP_DIM 4096
#define BATCH 2
#define TOKENS (BATCH * S_LEN)   // 4096

typedef float v8f __attribute__((ext_vector_type(8)));
typedef __bf16 bf16x16 __attribute__((ext_vector_type(16)));

union BFrag {            // one 16x32 (A) or 32x16 (B) bf16 WMMA operand
    uint4 q[2];          // 2 x 16B = 16 bf16 per lane
    bf16x16 v;
};
union BF4 { __bf16 h[4]; uint2 u; };

static __device__ __forceinline__ v8f wmma_bf16(const BFrag& a, const BFrag& b, v8f c) {
    return __builtin_amdgcn_wmma_f32_16x16x32_bf16(false, a.v, false, b.v,
                                                   (short)0, c, false, false);
}

// gfx1250 async global->LDS copy, 16B per lane, tracked by ASYNCcnt.
// ldsaddr: wave-relative LDS byte offset (low 32 bits of flat shared ptr).
static __device__ __forceinline__ void async_copy16(unsigned ldsaddr, const void* gptr) {
    unsigned long long ga = (unsigned long long)(uintptr_t)gptr;
    asm volatile("global_load_async_to_lds_b128 %0, %1, off"
                 :: "v"(ldsaddr), "v"(ga) : "memory");
}
static __device__ __forceinline__ void wait_async() {
    asm volatile("s_wait_asynccnt 0x0" ::: "memory");
}
static __device__ __forceinline__ unsigned lds_off(const void* p) {
    return (unsigned)(uintptr_t)p;   // LDS aperture: addr[31:0] = LDS offset
}

// ---------------------------------------------------------------------------
// Weight convert + transpose: src [K][N] f32 -> dst [N][K] bf16.
// ---------------------------------------------------------------------------
__global__ __launch_bounds__(256) void wconv_kernel(const float* __restrict__ src,
                                                    __bf16* __restrict__ dst,
                                                    int K, int N) {
    __shared__ float t[32][33];
    const int tc = threadIdx.x & 31;
    const int tr = threadIdx.x >> 5;
    const int k0 = blockIdx.y * 32;
    const int n0 = blockIdx.x * 32;
    #pragma unroll
    for (int i = 0; i < 4; ++i) {
        const int kk = tr + i * 8;
        t[kk][tc] = src[(size_t)(k0 + kk) * N + n0 + tc];
    }
    __syncthreads();
    #pragma unroll
    for (int i = 0; i < 4; ++i) {
        const int nn = tr + i * 8;
        dst[(size_t)(n0 + nn) * K + k0 + tc] = (__bf16)t[tc][nn];
    }
}

// ---------------------------------------------------------------------------
// LayerNorm: one block per token row (E=1024), f32 in -> bf16 out.
// ---------------------------------------------------------------------------
__global__ __launch_bounds__(256) void ln_kernel(const float* __restrict__ x,
                                                 const float* __restrict__ g,
                                                 const float* __restrict__ be,
                                                 __bf16* __restrict__ out) {
    const int row = blockIdx.x;
    const int tid = threadIdx.x;
    const float* xr = x + (size_t)row * E_DIM;
    float4 xv = ((const float4*)xr)[tid];

    float s = xv.x + xv.y + xv.z + xv.w;
    float s2 = xv.x * xv.x + xv.y * xv.y + xv.z * xv.z + xv.w * xv.w;
    #pragma unroll
    for (int m = 1; m < 32; m <<= 1) {
        s += __shfl_xor(s, m);
        s2 += __shfl_xor(s2, m);
    }
    __shared__ float rs[8], rs2[8];
    const int wv = tid >> 5;
    if ((tid & 31) == 0) { rs[wv] = s; rs2[wv] = s2; }
    __syncthreads();
    float ts = 0.f, ts2 = 0.f;
    #pragma unroll
    for (int i = 0; i < 8; ++i) { ts += rs[i]; ts2 += rs2[i]; }
    const float mu = ts * (1.0f / E_DIM);
    const float var = ts2 * (1.0f / E_DIM) - mu * mu;
    const float rstd = rsqrtf(var + 1e-5f);

    float4 gv = ((const float4*)g)[tid];
    float4 bv = ((const float4*)be)[tid];
    BF4 o;
    o.h[0] = (__bf16)((xv.x - mu) * rstd * gv.x + bv.x);
    o.h[1] = (__bf16)((xv.y - mu) * rstd * gv.y + bv.y);
    o.h[2] = (__bf16)((xv.z - mu) * rstd * gv.z + bv.z);
    o.h[3] = (__bf16)((xv.w - mu) * rstd * gv.w + bv.w);
    ((uint2*)(out + (size_t)row * E_DIM))[tid] = o.u;
}

// ---------------------------------------------------------------------------
// WMMA GEMM: C[M,N] = epilogue(A[M,K] @ Bt[N,K]^T + bias [+ resid])
// Block tile 128x128, BK=32, 8 waves (wave tile 32x64).
// Async-to-LDS double-buffered staging, one barrier per K-step.
// ---------------------------------------------------------------------------
enum { ACT_NONE = 0, ACT_GELU = 1 };
enum { OUT_F32_RES = 0, OUT_BF16 = 1, OUT_HEADS = 2, OUT_HEADS_T = 3 };

#define LDT 48   // LDS row stride in bf16 elems (96B, multiple of 16B)

template <int ACT, int OUTM, bool RES>
__global__ __launch_bounds__(256) void gemm_kernel(const __bf16* __restrict__ A,
                                                   const __bf16* __restrict__ Bt,
                                                   const float* __restrict__ bias,
                                                   const float* __restrict__ resid,
                                                   void* __restrict__ CoutV,
                                                   int M, int N, int K) {
    __shared__ __attribute__((aligned(16))) __bf16 As[2][128 * LDT];  // [m][k]
    __shared__ __attribute__((aligned(16))) __bf16 Bs[2][128 * LDT];  // [n][k]

    const int tid = threadIdx.x;
    const int wave = tid >> 5;
    const int lane = tid & 31;
    const int half = lane >> 4;
    const int lm = lane & 15;
    const int wm = wave >> 1;
    const int wn = wave & 1;
    const int Mbase = blockIdx.y * 128;
    const int Nbase = blockIdx.x * 128;

    // staging map: thread -> (row, k-half), same for A and B tiles
    const int st_r = tid >> 1;
    const int st_k = (tid & 1) * 16;
    const __bf16* gA = A + (size_t)(Mbase + st_r) * K + st_k;
    const __bf16* gB = Bt + (size_t)(Nbase + st_r) * K + st_k;

    auto stage = [&](int buf, int k0) {
        const unsigned la = lds_off(&As[buf][st_r * LDT + st_k]);
        const unsigned lb = lds_off(&Bs[buf][st_r * LDT + st_k]);
        async_copy16(la,      gA + k0);
        async_copy16(la + 16, gA + k0 + 8);
        async_copy16(lb,      gB + k0);
        async_copy16(lb + 16, gB + k0 + 8);
    };

    v8f acc[2][4];
    #pragma unroll
    for (int i = 0; i < 2; ++i)
        #pragma unroll
        for (int j = 0; j < 4; ++j)
            #pragma unroll
            for (int r = 0; r < 8; ++r) acc[i][j][r] = 0.0f;

    const int ksteps = K >> 5;
    stage(0, 0);
    wait_async();
    __syncthreads();

    for (int kb = 0; kb < ksteps; ++kb) {
        const int cur = kb & 1;
        if (kb + 1 < ksteps) stage(cur ^ 1, (kb + 1) << 5);   // overlap with WMMA

        BFrag af[2], bf_[4];
        #pragma unroll
        for (int mf = 0; mf < 2; ++mf) {
            const int base = (wm * 32 + mf * 16 + lm) * LDT;
            af[mf].q[0] = *(const uint4*)&As[cur][base + half * 16];
            af[mf].q[1] = *(const uint4*)&As[cur][base + half * 16 + 8];
        }
        #pragma unroll
        for (int nf = 0; nf < 4; ++nf) {
            const int base = (wn * 64 + nf * 16 + lm) * LDT + half * 16;
            bf_[nf].q[0] = *(const uint4*)&Bs[cur][base];
            bf_[nf].q[1] = *(const uint4*)&Bs[cur][base + 8];
        }
        #pragma unroll
        for (int mf = 0; mf < 2; ++mf)
            #pragma unroll
            for (int nf = 0; nf < 4; ++nf)
                acc[mf][nf] = wmma_bf16(af[mf], bf_[nf], acc[mf][nf]);

        wait_async();       // next buffer fully in LDS
        __syncthreads();    // all waves done with this buffer / see next one
    }

    // ---- epilogue ----
    float* Cf = (float*)CoutV;
    __bf16* Cb = (__bf16*)CoutV;
    #pragma unroll
    for (int mf = 0; mf < 2; ++mf) {
        #pragma unroll
        for (int nf = 0; nf < 4; ++nf) {
            const int gn = Nbase + wn * 64 + nf * 16 + lm;
            const int gm0 = Mbase + wm * 32 + mf * 16 + 8 * half;
            const float bvv = bias[gn];
            #pragma unroll
            for (int r = 0; r < 8; ++r) {
                const int gm = gm0 + r;
                float val = acc[mf][nf][r] + bvv;
                if (RES) val += resid[(size_t)gm * N + gn];
                if (ACT == ACT_GELU)
                    val = 0.5f * val * (1.0f + erff(val * 0.70710678118654752f));
                if (OUTM == OUT_F32_RES) {
                    Cf[(size_t)gm * N + gn] = val;
                } else if (OUTM == OUT_BF16) {
                    Cb[(size_t)gm * N + gn] = (__bf16)val;
                } else {
                    const int b = gm >> 11, sIdx = gm & 2047;
                    const int h = gn >> 6, d = gn & 63;
                    if (OUTM == OUT_HEADS)
                        Cb[(((size_t)b * H_NUM + h) * S_LEN + sIdx) * HD_DIM + d] = (__bf16)val;
                    else // OUT_HEADS_T: V stored [b][h][d][s]
                        Cb[(((size_t)b * H_NUM + h) * HD_DIM + d) * S_LEN + sIdx] = (__bf16)val;
                }
            }
        }
    }
}

// ---------------------------------------------------------------------------
// Flash attention. One block = one (b,h) x 128 Q rows; 8 waves x 16 rows.
// 64-key chunks, double-buffered async K/V staging, online softmax in regs.
// q,k: [b][h][s][64] bf16.  vt: [b][h][64][s] bf16.  ctx out: [b][s][E] bf16.
// ---------------------------------------------------------------------------
__global__ __launch_bounds__(256) void attn_kernel(const __bf16* __restrict__ q,
                                                   const __bf16* __restrict__ k,
                                                   const __bf16* __restrict__ vt,
                                                   __bf16* __restrict__ ctx) {
    __shared__ __attribute__((aligned(16))) __bf16 Ks[2][64 * 64];   // [key][d]
    __shared__ __attribute__((aligned(16))) __bf16 Vs[2][64 * 64];   // [d][key]
    __shared__ __attribute__((aligned(16))) __bf16 Ps[8 * 16 * 64];  // per-wave P

    const int tid = threadIdx.x;
    const int wave = tid >> 5;
    const int lane = tid & 31;
    const int half = lane >> 4;
    const int lm = lane & 15;

    const int bh = blockIdx.x >> 4;
    const int qt = blockIdx.x & 15;
    const int b = bh >> 4, h = bh & 15;
    const __bf16* qb = q + (size_t)bh * S_LEN * HD_DIM;
    const __bf16* kb = k + (size_t)bh * S_LEN * HD_DIM;
    const __bf16* vb = vt + (size_t)bh * HD_DIM * S_LEN;

    // Q A-fragments (16 rows x 64 d = two K=32 fragments) from global bf16
    const int qrow = qt * 128 + wave * 16 + lm;
    BFrag qa[2];
    #pragma unroll
    for (int c = 0; c < 2; ++c)
        #pragma unroll
        for (int g = 0; g < 2; ++g)
            qa[c].q[g] = *(const uint4*)(qb + (size_t)qrow * HD_DIM +
                                         c * 32 + half * 16 + g * 8);

    v8f accv[4];
    #pragma unroll
    for (int i = 0; i < 4; ++i)
        #pragma unroll
        for (int r = 0; r < 8; ++r) accv[i][r] = 0.0f;
    float mrow[8], lrow[8];
    #pragma unroll
    for (int r = 0; r < 8; ++r) { mrow[r] = -3.0e38f; lrow[r] = 0.0f; }

    const float scale = 0.125f;  // 1/sqrt(64)
    const int st_r = tid >> 2;            // key (for K) / d (for V)
    const int st_c = (tid & 3) * 16;      // 16-elem column offset

    auto stage = [&](int buf, int ks0) {
        const unsigned lk = lds_off(&Ks[buf][st_r * 64 + st_c]);
        const unsigned lv = lds_off(&Vs[buf][st_r * 64 + st_c]);
        async_copy16(lk,      kb + (size_t)(ks0 + st_r) * HD_DIM + st_c);
        async_copy16(lk + 16, kb + (size_t)(ks0 + st_r) * HD_DIM + st_c + 8);
        async_copy16(lv,      vb + (size_t)st_r * S_LEN + ks0 + st_c);
        async_copy16(lv + 16, vb + (size_t)st_r * S_LEN + ks0 + st_c + 8);
    };

    stage(0, 0);
    wait_async();
    __syncthreads();

    for (int it = 0; it < S_LEN / 64; ++it) {
        const int cur = it & 1;
        if (it + 1 < S_LEN / 64) stage(cur ^ 1, (it + 1) * 64);

        // scores: S(16x64) = Q(16x64) @ K^T(64x64) -> four 16x16 C frags
        v8f s[4];
        #pragma unroll
        for (int g = 0; g < 4; ++g)
            #pragma unroll
            for (int r = 0; r < 8; ++r) s[g][r] = 0.0f;
        #pragma unroll
        for (int g = 0; g < 4; ++g) {
            #pragma unroll
            for (int c = 0; c < 2; ++c) {
                BFrag kf;
                const int base = (g * 16 + lm) * 64 + c * 32 + half * 16;
                kf.q[0] = *(const uint4*)&Ks[cur][base];
                kf.q[1] = *(const uint4*)&Ks[cur][base + 8];
                s[g] = wmma_bf16(qa[c], kf, s[g]);
            }
        }

        // online softmax (rows r+8*half; reduce over the 16 lanes of a half)
        #pragma unroll
        for (int r = 0; r < 8; ++r) {
            float a0 = s[0][r] * scale, a1 = s[1][r] * scale;
            float a2 = s[2][r] * scale, a3 = s[3][r] * scale;
            float t = fmaxf(fmaxf(a0, a1), fmaxf(a2, a3));
            t = fmaxf(t, __shfl_xor(t, 1));
            t = fmaxf(t, __shfl_xor(t, 2));
            t = fmaxf(t, __shfl_xor(t, 4));
            t = fmaxf(t, __shfl_xor(t, 8));
            const float mn = fmaxf(mrow[r], t);
            const float alpha = __expf(mrow[r] - mn);
            mrow[r] = mn;
            const float p0 = __expf(a0 - mn);
            const float p1 = __expf(a1 - mn);
            const float p2 = __expf(a2 - mn);
            const float p3 = __expf(a3 - mn);
            float rsum = (p0 + p1) + (p2 + p3);
            rsum += __shfl_xor(rsum, 1);
            rsum += __shfl_xor(rsum, 2);
            rsum += __shfl_xor(rsum, 4);
            rsum += __shfl_xor(rsum, 8);
            lrow[r] = lrow[r] * alpha + rsum;
            #pragma unroll
            for (int c2 = 0; c2 < 4; ++c2) accv[c2][r] *= alpha;
            // bounce P through per-wave LDS to re-layout C -> A fragments
            const int prow = (wave * 16 + r + 8 * half) * 64;
            Ps[prow + lm] = (__bf16)p0;
            Ps[prow + 16 + lm] = (__bf16)p1;
            Ps[prow + 32 + lm] = (__bf16)p2;
            Ps[prow + 48 + lm] = (__bf16)p3;
        }

        BFrag pa0, pa1;   // P as two 16x32 A fragments (keys 0..31, 32..63)
        {
            const int prow = (wave * 16 + lm) * 64;
            pa0.q[0] = *(const uint4*)&Ps[prow + half * 16];
            pa0.q[1] = *(const uint4*)&Ps[prow + half * 16 + 8];
            pa1.q[0] = *(const uint4*)&Ps[prow + 32 + half * 16];
            pa1.q[1] = *(const uint4*)&Ps[prow + 32 + half * 16 + 8];
        }

        // ctx += P(16x64) @ V(64x64): four N=16 chunks x two key halves
        #pragma unroll
        for (int c2 = 0; c2 < 4; ++c2) {
            const int d = c2 * 16 + lm;
            BFrag vf;
            vf.q[0] = *(const uint4*)&Vs[cur][d * 64 + half * 16];
            vf.q[1] = *(const uint4*)&Vs[cur][d * 64 + half * 16 + 8];
            accv[c2] = wmma_bf16(pa0, vf, accv[c2]);
            vf.q[0] = *(const uint4*)&Vs[cur][d * 64 + 32 + half * 16];
            vf.q[1] = *(const uint4*)&Vs[cur][d * 64 + 32 + half * 16 + 8];
            accv[c2] = wmma_bf16(pa1, vf, accv[c2]);
        }

        wait_async();
        __syncthreads();
    }

    // write ctx rows back to [b][s][E] at column h*64 (bf16)
    #pragma unroll
    for (int c2 = 0; c2 < 4; ++c2) {
        #pragma unroll
        for (int r = 0; r < 8; ++r) {
            const int srow = qt * 128 + wave * 16 + r + 8 * half;
            const int e = h * 64 + c2 * 16 + lm;
            ctx[((size_t)b * S_LEN + srow) * E_DIM + e] = (__bf16)(accv[c2][r] / lrow[r]);
        }
    }
}

// ---------------------------------------------------------------------------
// Host-side orchestration.
// ---------------------------------------------------------------------------
extern "C" void kernel_launch(void* const* d_in, const int* in_sizes, int n_in,
                              void* d_out, int out_size, void* d_ws, size_t ws_size,
                              hipStream_t stream) {
    (void)in_sizes; (void)n_in; (void)out_size; (void)ws_size;
    const float* x   = (const float*)d_in[0];
    const float* wq  = (const float*)d_in[1];
    const float* bq  = (const float*)d_in[2];
    const float* wk  = (const float*)d_in[3];
    const float* bk  = (const float*)d_in[4];
    const float* wv  = (const float*)d_in[5];
    const float* bv  = (const float*)d_in[6];
    const float* wo  = (const float*)d_in[7];
    const float* bo  = (const float*)d_in[8];
    const float* w1  = (const float*)d_in[9];
    const float* b1  = (const float*)d_in[10];
    const float* w2  = (const float*)d_in[11];
    const float* b2  = (const float*)d_in[12];
    const float* g1  = (const float*)d_in[13];
    const float* be1 = (const float*)d_in[14];
    const float* g2  = (const float*)d_in[15];
    const float* be2 = (const float*)d_in[16];
    float* out = (float*)d_out;

    // workspace layout
    char* W = (char*)d_ws;
    const size_t TE = (size_t)TOKENS * E_DIM;           // 4M elems
    const size_t EE = (size_t)E_DIM * E_DIM;            // 1M elems
    const size_t EM = (size_t)E_DIM * MLP_DIM;          // 4M elems
    __bf16* wqT = (__bf16*)W; W += EE * 2;
    __bf16* wkT = (__bf16*)W; W += EE * 2;
    __bf16* wvT = (__bf16*)W; W += EE * 2;
    __bf16* woT = (__bf16*)W; W += EE * 2;
    __bf16* w1T = (__bf16*)W; W += EM * 2;
    __bf16* w2T = (__bf16*)W; W += EM * 2;
    __bf16* z0  = (__bf16*)W; W += TE * 2;
    __bf16* qB  = (__bf16*)W; W += TE * 2;
    __bf16* kB  = (__bf16*)W; W += TE * 2;
    __bf16* vT  = (__bf16*)W; W += TE * 2;
    __bf16* ctx = (__bf16*)W; W += TE * 2;
    __bf16* z1  = (__bf16*)W; W += TE * 2;
    __bf16* hB  = (__bf16*)W; W += (size_t)TOKENS * MLP_DIM * 2;
    float*  x1  = (float*)W;  W += TE * 4;

    const dim3 blk(256);
    const dim3 gE(E_DIM / 128, TOKENS / 128);           // (8, 32)
    const dim3 gM(MLP_DIM / 128, TOKENS / 128);         // (32, 32)

    // 0) convert + transpose all weights to bf16 [N][K]
    wconv_kernel<<<dim3(E_DIM / 32, E_DIM / 32), blk, 0, stream>>>(wq, wqT, E_DIM, E_DIM);
    wconv_kernel<<<dim3(E_DIM / 32, E_DIM / 32), blk, 0, stream>>>(wk, wkT, E_DIM, E_DIM);
    wconv_kernel<<<dim3(E_DIM / 32, E_DIM / 32), blk, 0, stream>>>(wv, wvT, E_DIM, E_DIM);
    wconv_kernel<<<dim3(E_DIM / 32, E_DIM / 32), blk, 0, stream>>>(wo, woT, E_DIM, E_DIM);
    wconv_kernel<<<dim3(MLP_DIM / 32, E_DIM / 32), blk, 0, stream>>>(w1, w1T, E_DIM, MLP_DIM);
    wconv_kernel<<<dim3(E_DIM / 32, MLP_DIM / 32), blk, 0, stream>>>(w2, w2T, MLP_DIM, E_DIM);

    // 1) z0 = LN1(x) (bf16)
    ln_kernel<<<TOKENS, blk, 0, stream>>>(x, g1, be1, z0);
    // 2) q,k,v projections (bf16, v stored transposed per head)
    gemm_kernel<ACT_NONE, OUT_HEADS, false><<<gE, blk, 0, stream>>>(z0, wqT, bq, nullptr, qB, TOKENS, E_DIM, E_DIM);
    gemm_kernel<ACT_NONE, OUT_HEADS, false><<<gE, blk, 0, stream>>>(z0, wkT, bk, nullptr, kB, TOKENS, E_DIM, E_DIM);
    gemm_kernel<ACT_NONE, OUT_HEADS_T, false><<<gE, blk, 0, stream>>>(z0, wvT, bv, nullptr, vT, TOKENS, E_DIM, E_DIM);
    // 3) attention -> ctx [b][s][E] bf16
    attn_kernel<<<BATCH * H_NUM * (S_LEN / 128), blk, 0, stream>>>(qB, kB, vT, ctx);
    // 4) x1 = x + ctx @ wo + bo  (f32)
    gemm_kernel<ACT_NONE, OUT_F32_RES, true><<<gE, blk, 0, stream>>>(ctx, woT, bo, x, x1, TOKENS, E_DIM, E_DIM);
    // 5) z1 = LN2(x1) (bf16)
    ln_kernel<<<TOKENS, blk, 0, stream>>>(x1, g2, be2, z1);
    // 6) h = gelu(z1 @ w1 + b1) (bf16)
    gemm_kernel<ACT_GELU, OUT_BF16, false><<<gM, blk, 0, stream>>>(z1, w1T, b1, nullptr, hB, TOKENS, MLP_DIM, E_DIM);
    // 7) out = x1 + h @ w2 + b2  (f32)
    gemm_kernel<ACT_NONE, OUT_F32_RES, true><<<gE, blk, 0, stream>>>(hB, w2T, b2, x1, out, TOKENS, E_DIM, MLP_DIM);
}
